// SwinTransformerBlock_42331197669436
// MI455X (gfx1250) — matrix operand order repaired and checked by reading, not compiled, not run on hardware
//
#include <hip/hip_runtime.h>

// ---------------------------------------------------------------------------
// Swin Transformer block, fused, f16 WMMA (gfx1250 / CDNA5, wave32)
// B=64 H=W=56 C=128 WS=7 SS=3 NH=4 HD=32 N=49 NW=64  -> 4096 windows, 200704 tokens
// ---------------------------------------------------------------------------

typedef __attribute__((ext_vector_type(16))) _Float16 v16h;
typedef __attribute__((ext_vector_type(8)))  _Float16 v8h;
typedef __attribute__((ext_vector_type(8)))  float    v8f;

#define SCALE 0.17677669529663687f   // 32^-0.5
#define LN_EPS 1e-3f

// Padded LDS row strides (in halves): stride_words % 32 == 4 -> the 16 rows
// touched by one fragment load hit 16 distinct banks instead of all hitting
// bank 0 (power-of-two strides are 16-way conflicted on the 64-bank LDS).
#define LDX  136   // xw / ctx tile   (cols 128)
#define LDQK 264   // Q|K tile        (cols 256)
#define LDV  72    // V^T per head    (cols 64)
#define LDP  72    // probs per head  (cols 64)
#define LDM  136   // mlp input tile  (cols 128)
#define LDH  520   // gelu hidden     (cols 512)

// ---- WMMA wrapper (probe-confirmed builtin) -------------------------------
__device__ __forceinline__ v8f wmma(v16h a, v16h b, v8f c) {
  return __builtin_amdgcn_wmma_f32_16x16x32_f16(false, a, false, b, (short)0, c,
                                                false, false);
}

// A fragment (16x32 f16, row-major source, ld in halves).
// ISA layout: lane L holds row M=L&15; lanes0-15: K = 0..7 (V0-3), 16..23 (V4-7);
// lanes16-31: K = 8..15, 24..31  => two contiguous 8-half (16B) chunks.
__device__ __forceinline__ v16h load_a(const _Float16* base, int row, int ld,
                                       int kbase, int hi) {
  const _Float16* p = base + row * ld + kbase + hi * 8;
  v16h a;
  *((v8h*)&a)     = *(const v8h*)(p);
  *((v8h*)&a + 1) = *(const v8h*)(p + 16);
  return a;
}

// B fragment (32x16 f16). Source stored K-contiguous per column N (i.e. [N][K]).
// lane L holds column N=L&15; lanes0-15: K=0..7,16..23, lanes16-31: K=8..15,24..31.
// Two 16B loads (16B alignment is all the padded strides guarantee).
__device__ __forceinline__ v16h load_b(const _Float16* rowbase, int kbase, int hi) {
  const _Float16* p = rowbase + kbase + hi * 8;
  v16h b;
  *((v8h*)&b)     = *(const v8h*)(p);
  *((v8h*)&b + 1) = *(const v8h*)(p + 16);
  return b;
}

// ---- CDNA5 async global->LDS copy (ASYNCcnt path, no VGPR round-trip) -----
__device__ __forceinline__ void async_ld16(unsigned lds_off, const void* gaddr) {
  asm volatile("global_load_async_to_lds_b128 %0, %1, off"
               :: "v"(lds_off), "v"(gaddr) : "memory");
}
__device__ __forceinline__ void async_wait0() {
  asm volatile("s_wait_asynccnt 0" ::: "memory");
}

// ---- weight convert + transpose: f32 [K][N] -> f16 [N][K] -----------------
__global__ void k_t16(const float* __restrict__ src, _Float16* __restrict__ dst,
                      int K, int N) {
  int i = blockIdx.x * 256 + threadIdx.x;
  if (i >= K * N) return;
  int k = i / N, n = i - k * N;
  dst[n * K + k] = (_Float16)src[i];
}

// ---- LayerNorm over 128 channels, one wave per token ----------------------
__device__ __forceinline__ void ln_row(const float* __restrict__ src,
                                       const float* __restrict__ g,
                                       const float* __restrict__ be,
                                       _Float16* __restrict__ dst, int lane) {
  float4 v = *(const float4*)(src + lane * 4);
  float s = v.x + v.y + v.z + v.w;
#pragma unroll
  for (int m = 1; m <= 16; m <<= 1) s += __shfl_xor(s, m, 32);
  float mu = s * (1.0f / 128.0f);
  float d0 = v.x - mu, d1 = v.y - mu, d2 = v.z - mu, d3 = v.w - mu;
  float q = d0 * d0 + d1 * d1 + d2 * d2 + d3 * d3;
#pragma unroll
  for (int m = 1; m <= 16; m <<= 1) q += __shfl_xor(q, m, 32);
  float rstd = rsqrtf(q * (1.0f / 128.0f) + LN_EPS);
  int c = lane * 4;
  dst[c + 0] = (_Float16)(d0 * rstd * g[c + 0] + be[c + 0]);
  dst[c + 1] = (_Float16)(d1 * rstd * g[c + 1] + be[c + 1]);
  dst[c + 2] = (_Float16)(d2 * rstd * g[c + 2] + be[c + 2]);
  dst[c + 3] = (_Float16)(d3 * rstd * g[c + 3] + be[c + 3]);
}

// LN1 + cyclic shift(-3,-3) + window partition + pad rows 49..63 with zeros.
// Output xw16: [4096 windows][64 rows][128] f16. grid = 32768 x 256.
__global__ void k_ln1_win(const float* __restrict__ x, const float* __restrict__ g,
                          const float* __restrict__ be, _Float16* __restrict__ xw) {
  int wv = threadIdx.x >> 5, lane = threadIdx.x & 31;
  int R = blockIdx.x * 8 + wv;                 // 0..262143
  int win = R >> 6, n = R & 63;
  _Float16* dst = xw + (size_t)R * 128;
  if (n >= 49) {                               // zero pad row (uniform per wave)
    *(uint2*)(dst + lane * 4) = make_uint2(0u, 0u);
    return;
  }
  int b = win >> 6, w64 = win & 63, wr = w64 >> 3, wc = w64 & 7;
  int hs = wr * 7 + n / 7, vs = wc * 7 + n % 7;
  int h = hs + 3;  if (h  >= 56) h  -= 56;     // roll(-3): src = (dst+3)%56
  int wp = vs + 3; if (wp >= 56) wp -= 56;
  const float* src = x + ((size_t)b * 3136 + h * 56 + wp) * 128;
  ln_row(src, g, be, dst, lane);
}

// LN2: y (d_out) -> m16 [200704][128] f16. grid = 25088 x 256.
__global__ void k_ln2(const float* __restrict__ y, const float* __restrict__ g,
                      const float* __restrict__ be, _Float16* __restrict__ m16) {
  int wv = threadIdx.x >> 5, lane = threadIdx.x & 31;
  int R = blockIdx.x * 8 + wv;                 // 0..200703
  ln_row(y + (size_t)R * 128, g, be, m16 + (size_t)R * 128, lane);
}

// ---- fused per-window attention: QKV -> scores -> softmax -> @V -> proj ---
// one workgroup (8 waves) per window; ~104KB LDS
__global__ __launch_bounds__(256) void k_attn(
    const _Float16* __restrict__ xw, const _Float16* __restrict__ WqkvT,
    const float* __restrict__ bqkv, const _Float16* __restrict__ WprojT,
    const float* __restrict__ bproj, const float* __restrict__ relb,
    const float* __restrict__ x, float* __restrict__ y) {
  __shared__ __align__(32) _Float16 ls_xw[64 * LDX];      // reused as ctx
  __shared__ __align__(32) _Float16 ls_qk[64 * LDQK];     // Q cols 0..127, K cols 128..255
  __shared__ __align__(32) _Float16 ls_vT[4 * 32 * LDV];  // V^T [head][d][key]
  __shared__ __align__(32) _Float16 ls_p [4 * 64 * LDP];  // probs [head][n][m]

  int tid = threadIdx.x, wv = tid >> 5, lane = tid & 31;
  int hi = lane >> 4, ln15 = lane & 15;
  int win = blockIdx.x, b = win >> 6, w64 = win & 63, wr = w64 >> 3, wc = w64 & 7;

  // stage input tile via async L2->LDS DMA, row-remapped into padded layout
  {
    const _Float16* src = xw + (size_t)win * 8192;
    for (int i = tid; i < 1024; i += 256) {            // 1024 x 16B
      int row = i >> 4, seg = i & 15;
      async_ld16((unsigned)(size_t)(ls_xw + row * LDX + seg * 8),
                 src + row * 128 + seg * 8);
    }
    async_wait0();
  }
  __syncthreads();

  // ---- QKV GEMM: 64x384 = 64x128 @ 128x384 ; 96 tiles, 4 K-steps ----
  for (int t = wv; t < 96; t += 8) {
    int ntile = t >> 2, mtile = t & 3;                 // mtile wave-invariant
    v8f acc = {0, 0, 0, 0, 0, 0, 0, 0};
#pragma unroll
    for (int k = 0; k < 4; ++k) {
      v16h a = load_a(ls_xw, mtile * 16 + ln15, LDX, k * 32, hi);
      v16h bb = load_b(WqkvT + (ntile * 16 + ln15) * 128, k * 32, hi);
      acc = wmma(a, bb, acc);
    }
    int col = ntile * 16 + ln15;
    float bias = bqkv[col];
    if (ntile < 16) {                          // Q or K: row-major store
#pragma unroll
      for (int r = 0; r < 8; ++r) {
        int row = mtile * 16 + r + hi * 8;
        ls_qk[row * LDQK + col] = (_Float16)(acc[r] + bias);
      }
    } else {                                   // V: transposed per head
      int hv = (ntile - 16) >> 1;
      int dsub = ((ntile - 16) & 1) * 16 + ln15;
#pragma unroll
      for (int r = 0; r < 8; ++r) {
        int row = mtile * 16 + r + hi * 8;
        ls_vT[(hv * 32 + dsub) * LDV + row] = (_Float16)(acc[r] + bias);
      }
    }
  }
  __syncthreads();

  // ---- scores + bias + mask + softmax : 16 (head,mtile) tasks ----
  for (int t = wv; t < 16; t += 8) {
    int head = t >> 2, mtile = t & 3;
    v8f acc[4];
#pragma unroll
    for (int j = 0; j < 4; ++j) acc[j] = (v8f){0, 0, 0, 0, 0, 0, 0, 0};
    v16h a = load_a(ls_qk, mtile * 16 + ln15, LDQK, head * 32, hi); // Q, K-dim=HD=32
#pragma unroll
    for (int j = 0; j < 4; ++j) {              // key tiles; B from K rows (d-contig)
      v16h bb = load_b(ls_qk + (j * 16 + ln15) * LDQK + 128 + head * 32, 0, hi);
      acc[j] = wmma(a, bb, acc[j]);
    }
    // elementwise: scale + rel-pos bias + shift mask; pad -> -1e30
#pragma unroll
    for (int j = 0; j < 4; ++j) {
      int m = j * 16 + ln15;
      int rm = m / 7, cm = m % 7;
      int hm = wr * 7 + rm, wm = wc * 7 + cm;
      int regm = (hm < 49 ? 0 : (hm < 53 ? 1 : 2)) * 3 + (wm < 49 ? 0 : (wm < 53 ? 1 : 2));
#pragma unroll
      for (int r = 0; r < 8; ++r) {
        int n = mtile * 16 + r + hi * 8;
        float s;
        if (m >= 49 || n >= 49) {
          s = -1e30f;
        } else {
          int rn = n / 7, cn = n % 7;
          int hn = wr * 7 + rn, wn = wc * 7 + cn;
          int regn = (hn < 49 ? 0 : (hn < 53 ? 1 : 2)) * 3 +
                     (wn < 49 ? 0 : (wn < 53 ? 1 : 2));
          int idx = (rn - rm + 6) * 13 + (cn - cm + 6);
          s = acc[j][r] * SCALE + relb[idx * 4 + head];
          if (regn != regm) s -= 100.0f;
        }
        acc[j][r] = s;
      }
    }
    // softmax per row: a row lives in one 16-lane half-wave across the 4 tiles
#pragma unroll
    for (int r = 0; r < 8; ++r) {
      float mx = fmaxf(fmaxf(acc[0][r], acc[1][r]), fmaxf(acc[2][r], acc[3][r]));
#pragma unroll
      for (int msk = 1; msk <= 8; msk <<= 1) mx = fmaxf(mx, __shfl_xor(mx, msk, 32));
      float e0 = __expf(acc[0][r] - mx), e1 = __expf(acc[1][r] - mx);
      float e2 = __expf(acc[2][r] - mx), e3 = __expf(acc[3][r] - mx);
      float sm = e0 + e1 + e2 + e3;
#pragma unroll
      for (int msk = 1; msk <= 8; msk <<= 1) sm += __shfl_xor(sm, msk, 32);
      float inv = 1.0f / sm;
      int n = mtile * 16 + r + hi * 8;
      _Float16* prow = ls_p + (head * 64 + n) * LDP;
      prow[ln15 +  0] = (_Float16)(e0 * inv);
      prow[ln15 + 16] = (_Float16)(e1 * inv);
      prow[ln15 + 32] = (_Float16)(e2 * inv);
      prow[ln15 + 48] = (_Float16)(e3 * inv);
    }
  }
  __syncthreads();

  // ---- attn @ V : 32 tiles (head, mtile, dtile), K=64 -> 2 steps ----
  for (int t = wv; t < 32; t += 8) {
    int head = t >> 3, rem = t & 7, mtile = rem >> 1, ntile = rem & 1;
    v8f acc = {0, 0, 0, 0, 0, 0, 0, 0};
#pragma unroll
    for (int k = 0; k < 2; ++k) {
      v16h a = load_a(ls_p + head * 64 * LDP, mtile * 16 + ln15, LDP, k * 32, hi);
      v16h bb = load_b(ls_vT + (head * 32 + ntile * 16 + ln15) * LDV, k * 32, hi);
      acc = wmma(a, bb, acc);
    }
    int col = head * 32 + ntile * 16 + ln15;
#pragma unroll
    for (int r = 0; r < 8; ++r) {
      int row = mtile * 16 + r + hi * 8;
      ls_xw[row * LDX + col] = (_Float16)acc[r];   // ctx overlays dead xw tile
    }
  }
  __syncthreads();

  // ---- proj: 64x128 = 64x128 @ 128x128 ; write residual to d_out ----
  for (int t = wv; t < 32; t += 8) {
    int mtile = t >> 3, ntile = t & 7;
    v8f acc = {0, 0, 0, 0, 0, 0, 0, 0};
#pragma unroll
    for (int k = 0; k < 4; ++k) {
      v16h a = load_a(ls_xw, mtile * 16 + ln15, LDX, k * 32, hi);
      v16h bb = load_b(WprojT + (ntile * 16 + ln15) * 128, k * 32, hi);
      acc = wmma(a, bb, acc);
    }
    int col = ntile * 16 + ln15;
    float bias = bproj[col];
#pragma unroll
    for (int r = 0; r < 8; ++r) {
      int tokr = mtile * 16 + r + hi * 8;
      if (tokr < 49) {                         // window reverse + roll(+3,+3)
        int rn = tokr / 7, cn = tokr % 7;
        int hh = wr * 7 + rn + 3;  if (hh >= 56) hh -= 56;
        int ww2 = wc * 7 + cn + 3; if (ww2 >= 56) ww2 -= 56;
        size_t addr = ((size_t)b * 3136 + hh * 56 + ww2) * 128 + col;
        y[addr] = x[addr] + acc[r] + bias;     // shortcut residual
      }
    }
  }
}

// ---- fused MLP per 64-token tile: fc1 + exact GELU + fc2, d_out += m ------
__global__ __launch_bounds__(256) void k_mlp(
    const _Float16* __restrict__ m16, const _Float16* __restrict__ W1T,
    const float* __restrict__ b1, const _Float16* __restrict__ W2T,
    const float* __restrict__ b2, float* __restrict__ y) {
  __shared__ __align__(32) _Float16 ls_m[64 * LDM];
  __shared__ __align__(32) _Float16 ls_h[64 * LDH];
  int tid = threadIdx.x, wv = tid >> 5, lane = tid & 31;
  int hi = lane >> 4, ln15 = lane & 15;

  { // stage LN2'd tokens via async L2->LDS DMA into padded layout
    const _Float16* src = m16 + (size_t)blockIdx.x * 8192;
    for (int i = tid; i < 1024; i += 256) {
      int row = i >> 4, seg = i & 15;
      async_ld16((unsigned)(size_t)(ls_m + row * LDM + seg * 8),
                 src + row * 128 + seg * 8);
    }
    async_wait0();
  }
  __syncthreads();

  // fc1: 64x512 = 64x128 @ 128x512 ; 128 tiles
  for (int t = wv; t < 128; t += 8) {
    int ntile = t >> 2, mtile = t & 3;
    v8f acc = {0, 0, 0, 0, 0, 0, 0, 0};
#pragma unroll
    for (int k = 0; k < 4; ++k) {
      v16h a = load_a(ls_m, mtile * 16 + ln15, LDM, k * 32, hi);
      v16h bb = load_b(W1T + (ntile * 16 + ln15) * 128, k * 32, hi);
      acc = wmma(a, bb, acc);
    }
    int col = ntile * 16 + ln15;
    float bias = b1[col];
#pragma unroll
    for (int r = 0; r < 8; ++r) {
      int row = mtile * 16 + r + hi * 8;
      float v = acc[r] + bias;
      float gl = 0.5f * v * (1.0f + erff(v * 0.70710678118654752f)); // exact GELU
      ls_h[row * LDH + col] = (_Float16)gl;
    }
  }
  __syncthreads();

  // fc2: 64x128 = 64x512 @ 512x128 ; 32 tiles, 16 K-steps
  for (int t = wv; t < 32; t += 8) {
    int mtile = t >> 3, ntile = t & 7;
    v8f acc = {0, 0, 0, 0, 0, 0, 0, 0};
#pragma unroll
    for (int k = 0; k < 16; ++k) {
      v16h a = load_a(ls_h, mtile * 16 + ln15, LDH, k * 32, hi);
      v16h bb = load_b(W2T + (ntile * 16 + ln15) * 512, k * 32, hi);
      acc = wmma(a, bb, acc);
    }
    int col = ntile * 16 + ln15;
    float bias = b2[col];
#pragma unroll
    for (int r = 0; r < 8; ++r) {
      int row = mtile * 16 + r + hi * 8;
      size_t addr = ((size_t)blockIdx.x * 64 + row) * 128 + col;
      y[addr] = y[addr] + acc[r] + bias;       // final residual, in place
    }
  }
}

// ---------------------------------------------------------------------------
extern "C" void kernel_launch(void* const* d_in, const int* in_sizes, int n_in,
                              void* d_out, int out_size, void* d_ws, size_t ws_size,
                              hipStream_t stream) {
  (void)in_sizes; (void)n_in; (void)out_size; (void)ws_size;
  const float* x     = (const float*)d_in[0];
  const float* g1    = (const float*)d_in[1];
  const float* be1   = (const float*)d_in[2];
  const float* Wqkv  = (const float*)d_in[3];
  const float* bqkv  = (const float*)d_in[4];
  const float* relb  = (const float*)d_in[5];
  const float* Wproj = (const float*)d_in[6];
  const float* bproj = (const float*)d_in[7];
  const float* g2    = (const float*)d_in[8];
  const float* be2   = (const float*)d_in[9];
  const float* Wfc1  = (const float*)d_in[10];
  const float* bfc1  = (const float*)d_in[11];
  const float* Wfc2  = (const float*)d_in[12];
  const float* bfc2  = (const float*)d_in[13];
  float* y = (float*)d_out;

  char* ws = (char*)d_ws;                       // ~113.4 MB used
  _Float16* WqkvT  = (_Float16*)(ws + 0);            // 384x128  (98304 B)
  _Float16* WprojT = (_Float16*)(ws + 98304);        // 128x128  (32768 B)
  _Float16* W1T    = (_Float16*)(ws + 131072);       // 512x128  (131072 B)
  _Float16* W2T    = (_Float16*)(ws + 262144);       // 128x512  (131072 B)
  _Float16* xw16   = (_Float16*)(ws + 393216);       // 4096x64x128 (64 MB)
  _Float16* m16    = (_Float16*)(ws + 393216 + 67108864); // 200704x128 (49 MB)

  k_t16<<<(49152 + 255) / 256, 256, 0, stream>>>(Wqkv,  WqkvT,  128, 384);
  k_t16<<<(16384 + 255) / 256, 256, 0, stream>>>(Wproj, WprojT, 128, 128);
  k_t16<<<(65536 + 255) / 256, 256, 0, stream>>>(Wfc1,  W1T,    128, 512);
  k_t16<<<(65536 + 255) / 256, 256, 0, stream>>>(Wfc2,  W2T,    512, 128);

  k_ln1_win<<<32768, 256, 0, stream>>>(x, g1, be1, xw16);
  k_attn<<<4096, 256, 0, stream>>>(xw16, WqkvT, bqkv, WprojT, bproj, relb, x, y);
  k_ln2<<<25088, 256, 0, stream>>>(y, g2, be2, m16);
  k_mlp<<<3136, 256, 0, stream>>>(m16, W1T, bfc1, W2T, bfc2, y);
}
